// VideoAttentionBlock_28166395527965
// MI455X (gfx1250) — matrix operand order repaired and checked by reading, not compiled, not run on hardware
//
#include <hip/hip_runtime.h>
#include <hip/hip_bf16.h>

// VideoAttentionBlock for MI455X (gfx1250, wave32, WMMA bf16, async LDS staging).
// N=4096 tokens, C=512, 8 heads x 64 head-dim.

#define NTOK 4096
#define CDIM 512
#define NH   8
#define HD   64

typedef __attribute__((ext_vector_type(16))) __bf16 v16bf;
typedef __attribute__((ext_vector_type(8)))  float  v8f;

union Frag16 {
  v16bf v;
  unsigned int u[8];
  uint4 q[2];
};

__device__ __forceinline__ unsigned short f32_to_bf16(float f) {
  unsigned int x = __float_as_uint(f);
  unsigned int r = 0x7fffu + ((x >> 16) & 1u);   // round-to-nearest-even
  return (unsigned short)((x + r) >> 16);
}

// ---- CDNA5 async copy: 16B global -> LDS per lane, tracked by ASYNCcnt ----
__device__ __forceinline__ void async_copy16(const unsigned short* g, unsigned short* l) {
  unsigned laddr = (unsigned)(unsigned long long)
      (__attribute__((address_space(3))) unsigned short*)l;
  unsigned long long gaddr = (unsigned long long)g;
  asm volatile("global_load_async_to_lds_b128 %0, %1, off"
               :: "v"(laddr), "v"(gaddr) : "memory");
}

__device__ __forceinline__ void wait_async() {
#if __has_builtin(__builtin_amdgcn_s_wait_asynccnt)
  __builtin_amdgcn_s_wait_asynccnt(0);
#else
  asm volatile("s_wait_asynccnt 0x0" ::: "memory");
#endif
}

// A-matrix fragment (16x32 bf16, M x K), row-major source with leading dim ld.
// Layout: lane m = lane&15; lanes 0-15 hold K=0..15, lanes 16-31 K=16..31.
__device__ __forceinline__ Frag16 load_a_frag(const unsigned short* base, int ld, int lane) {
  int m  = lane & 15;
  int kb = (lane >> 4) << 4;          // 0 or 16
  const unsigned short* p = base + m * ld + kb;
  Frag16 f;
  f.q[0] = *reinterpret_cast<const uint4*>(p);        // K = kb+0..kb+7
  f.q[1] = *reinterpret_cast<const uint4*>(p + 8);    // K = kb+8..kb+15
  return f;
}

// B-matrix fragment (32x16 bf16, K x N) loaded from B^T stored [N][K] row-major.
// Layout (same k order as A): lane n = lane&15; lanes 0-15 hold K=0..15, lanes 16-31 K=16..31.
__device__ __forceinline__ Frag16 load_bt_frag(const unsigned short* base, int ld, int lane) {
  int n  = lane & 15;
  int kb = (lane >> 4) << 4;          // 0 or 16
  const unsigned short* p = base + n * ld + kb;
  Frag16 f;
  f.q[0] = *reinterpret_cast<const uint4*>(p);
  f.q[1] = *reinterpret_cast<const uint4*>(p + 8);
  return f;
}

__device__ __forceinline__ v8f wmma_bf16(Frag16 a, Frag16 b, v8f c) {
  return __builtin_amdgcn_wmma_f32_16x16x32_bf16(false, a.v, false, b.v,
                                                 (short)0, c, false, false);
}

// ---------------------------------------------------------------- weights -> bf16, transposed [N][K]
__global__ void __launch_bounds__(256) k_w_bf16_t(const float* __restrict__ W,
                                                  unsigned short* __restrict__ WT,
                                                  int K, int Nn) {
  int i = blockIdx.x * 256 + threadIdx.x;
  if (i >= K * Nn) return;
  int k = i / Nn, n = i - k * Nn;
  WT[n * K + k] = f32_to_bf16(W[i]);
}

// ---------------------------------------------------------------- LayerNorm -> bf16
__global__ void __launch_bounds__(256) k_layernorm(const float* __restrict__ X,
                                                   const float* __restrict__ g,
                                                   const float* __restrict__ b,
                                                   unsigned short* __restrict__ H) {
  __shared__ float red[256];
  int r = blockIdx.x, t = threadIdx.x;
  const float* xr = X + r * CDIM;
  float v0 = xr[t], v1 = xr[t + 256];
  red[t] = v0 + v1;
  __syncthreads();
  for (int off = 128; off > 0; off >>= 1) {
    if (t < off) red[t] += red[t + off];
    __syncthreads();
  }
  float mu = red[0] * (1.0f / CDIM);
  __syncthreads();
  float d0 = v0 - mu, d1 = v1 - mu;
  red[t] = d0 * d0 + d1 * d1;
  __syncthreads();
  for (int off = 128; off > 0; off >>= 1) {
    if (t < off) red[t] += red[t + off];
    __syncthreads();
  }
  float rstd = rsqrtf(red[0] * (1.0f / CDIM) + 1e-5f);
  H[r * CDIM + t]       = f32_to_bf16(d0 * rstd * g[t] + b[t]);
  H[r * CDIM + t + 256] = f32_to_bf16(d1 * rstd * g[t + 256] + b[t + 256]);
}

// Stage one 64x64 B^T tile (rows n0..n0+63, k chunk of 64) into LDS: 8KB, 2x16B/thread.
__device__ __forceinline__ void stage_bt_tile64(const unsigned short* BTrow, int ldb,
                                                unsigned short* buf, int kc, int tid) {
  int r = tid >> 3, c = tid & 7;            // 32 rows x 8 chunks, twice
  async_copy16(BTrow + r * ldb + kc + c * 8,        buf + r * 64 + c * 8);
  async_copy16(BTrow + (r + 32) * ldb + kc + c * 8, buf + (r + 32) * 64 + c * 8);
}

// ---------------------------------------------------------------- QKV GEMM: [4096x512] x [512x1536]
// Block = 8 waves = 128 rows x 64 cols; 64-deep B strip staged async to LDS.
__global__ void __launch_bounds__(256) k_gemm_qkv(const unsigned short* __restrict__ A,
                                                  const unsigned short* __restrict__ BT,
                                                  float* __restrict__ C) {
  __shared__ __align__(16) unsigned short sB[2][64 * 64];
  int tid = threadIdx.x, lane = tid & 31, wv = tid >> 5;
  int sup = blockIdx.x / 24;                // 0..31  (128-row supertile)
  int nt  = blockIdx.x % 24;                // 64-col tile
  int m0 = sup * 128 + wv * 16, n0 = nt * 64;
  const unsigned short* Arow  = A + m0 * CDIM;
  const unsigned short* BTrow = BT + n0 * CDIM;
  v8f acc[4] = {};
  stage_bt_tile64(BTrow, CDIM, sB[0], 0, tid);
  for (int k0 = 0, it = 0; k0 < CDIM; k0 += 64, ++it) {
    int cur = it & 1;
    wait_async();
    __syncthreads();
    if (k0 + 64 < CDIM) stage_bt_tile64(BTrow, CDIM, sB[cur ^ 1], k0 + 64, tid);
    Frag16 af0 = load_a_frag(Arow + k0,      CDIM, lane);
    Frag16 af1 = load_a_frag(Arow + k0 + 32, CDIM, lane);
#pragma unroll
    for (int s = 0; s < 4; ++s) {
      acc[s] = wmma_bf16(af0, load_bt_frag(sB[cur] + s * 16 * 64,      64, lane), acc[s]);
      acc[s] = wmma_bf16(af1, load_bt_frag(sB[cur] + s * 16 * 64 + 32, 64, lane), acc[s]);
    }
    __syncthreads();
  }
  int n = lane & 15, hh = lane >> 4;
#pragma unroll
  for (int s = 0; s < 4; ++s)
#pragma unroll
    for (int i = 0; i < 8; ++i)
      C[(m0 + hh * 8 + i) * (3 * CDIM) + n0 + s * 16 + n] = acc[s][i];
}

// ---------------------------------------------------------------- RoPE + head split (q,k bf16 [Nh][N][D]; v^T bf16 [Nh][D][N])
__global__ void __launch_bounds__(256) k_rope_split(const float* __restrict__ qkv,
                                                    const float* __restrict__ fr,
                                                    unsigned short* __restrict__ Qo,
                                                    unsigned short* __restrict__ Ko,
                                                    unsigned short* __restrict__ VTo) {
  int idx = blockIdx.x * 256 + threadIdx.x;     // N * NH * 32 threads
  int n = idx >> 8;
  int head = (idx >> 5) & 7;
  int j = idx & 31;
  const float* row = qkv + n * (3 * CDIM);
  float c = fr[n * HD + 2 * j];
  float s = fr[n * HD + 2 * j + 1];
  int off = head * HD + 2 * j;
  float qa = row[off],            qb = row[off + 1];
  float ka = row[CDIM + off],     kb = row[CDIM + off + 1];
  float va = row[2 * CDIM + off], vb = row[2 * CDIM + off + 1];
  int base = head * (NTOK * HD) + n * HD + 2 * j;
  Qo[base]     = f32_to_bf16(qa * c - qb * s);
  Qo[base + 1] = f32_to_bf16(qa * s + qb * c);
  Ko[base]     = f32_to_bf16(ka * c - kb * s);
  Ko[base + 1] = f32_to_bf16(ka * s + kb * c);
  int vbase = head * (HD * NTOK) + (2 * j) * NTOK + n;
  VTo[vbase]        = f32_to_bf16(va);
  VTo[vbase + NTOK] = f32_to_bf16(vb);
}

// Build P^T B-fragment (32 keys x 16 queries) from two exp'd S^T C/D reg arrays;
// half-wave shfl_xor supplies the other 8-key slab per query column.
__device__ __forceinline__ Frag16 build_p_frag(const float* p0, const float* p1, bool hi) {
  float x0[8], x1[8];
#pragma unroll
  for (int i = 0; i < 8; ++i) { x0[i] = __shfl_xor(p0[i], 16); x1[i] = __shfl_xor(p1[i], 16); }
  Frag16 bp;
#pragma unroll
  for (int w2 = 0; w2 < 8; ++w2) {
    const int e0 = 2 * w2, e1 = 2 * w2 + 1;
    float f0 = (e0 < 8) ? (hi ? x1[e0] : p0[e0]) : (hi ? p1[e0 - 8] : x0[e0 - 8]);
    float f1 = (e1 < 8) ? (hi ? x1[e1] : p0[e1]) : (hi ? p1[e1 - 8] : x0[e1 - 8]);
    bp.u[w2] = (unsigned int)f32_to_bf16(f0) | ((unsigned int)f32_to_bf16(f1) << 16);
  }
  return bp;
}

// ---------------------------------------------------------------- flash attention
// Block = 1 head x 128 queries (8 waves x 16). 64-key K/V^T tiles staged async
// into LDS, shared by all 8 waves; double-buffered; one softmax update / 64 keys.
__global__ void __launch_bounds__(256) k_attn(const unsigned short* __restrict__ Q,
                                              const unsigned short* __restrict__ K,
                                              const unsigned short* __restrict__ VT,
                                              unsigned short* __restrict__ O) {
  __shared__ __align__(16) unsigned short sK[2][64 * HD];   // [key][d]
  __shared__ __align__(16) unsigned short sV[2][HD * 64];   // [d][key]
  const int tid = threadIdx.x;
  const int lane = tid & 31;
  const int head = blockIdx.x >> 5;                 // 0..7
  const int q0 = (blockIdx.x & 31) * 128 + (tid >> 5) * 16;
  const unsigned short* Qh = Q  + head * (NTOK * HD);
  const unsigned short* Kh = K  + head * (NTOK * HD);
  const unsigned short* Vh = VT + head * (HD * NTOK);
  const float scale = 0.125f;   // 1/sqrt(64)

  // Q^T B-fragments are invariant across the key loop.
  Frag16 bq0 = load_bt_frag(Qh + q0 * HD,      HD, lane);
  Frag16 bq1 = load_bt_frag(Qh + q0 * HD + 32, HD, lane);

  v8f o0 = {}, o1 = {}, o2 = {}, o3 = {};      // O^T tiles, d0 = 0,16,32,48
  float m_run = -3.0e30f, l_run = 0.0f;

  // Stage a 64-key tile: K tile is 8KB contiguous; V^T tile rows are 128B runs.
  {
    async_copy16(Kh + tid * 8,         sK[0] + tid * 8);
    async_copy16(Kh + (tid + 256) * 8, sK[0] + (tid + 256) * 8);
    int r = tid >> 3, c = tid & 7;
    async_copy16(Vh + r * NTOK + c * 8,        sV[0] + r * 64 + c * 8);
    async_copy16(Vh + (r + 32) * NTOK + c * 8, sV[0] + (r + 32) * 64 + c * 8);
  }

  for (int kc = 0, it = 0; kc < NTOK; kc += 64, ++it) {
    const int cur = it & 1;
    wait_async();
    __syncthreads();
    if (kc + 64 < NTOK) {
      const int kn = kc + 64;
      async_copy16(Kh + kn * HD + tid * 8,         sK[cur ^ 1] + tid * 8);
      async_copy16(Kh + kn * HD + (tid + 256) * 8, sK[cur ^ 1] + (tid + 256) * 8);
      int r = tid >> 3, c = tid & 7;
      async_copy16(Vh + r * NTOK + kn + c * 8,        sV[cur ^ 1] + r * 64 + c * 8);
      async_copy16(Vh + (r + 32) * NTOK + kn + c * 8, sV[cur ^ 1] + (r + 32) * 64 + c * 8);
    }
    const unsigned short* kt = sK[cur];
    const unsigned short* vt = sV[cur];

    // S^T tiles for 64 keys: output M=key, N=query.
    v8f s[4] = {};
#pragma unroll
    for (int j = 0; j < 4; ++j) {
      s[j] = wmma_bf16(load_a_frag(kt + (j * 16) * HD,      HD, lane), bq0, s[j]);
      s[j] = wmma_bf16(load_a_frag(kt + (j * 16) * HD + 32, HD, lane), bq1, s[j]);
    }

    // One online-softmax update per 64 keys.
    float sc[4][8];
    float tmax = -3.0e30f;
#pragma unroll
    for (int j = 0; j < 4; ++j)
#pragma unroll
      for (int i = 0; i < 8; ++i) {
        sc[j][i] = s[j][i] * scale;
        tmax = fmaxf(tmax, sc[j][i]);
      }
    tmax = fmaxf(tmax, __shfl_xor(tmax, 16));
    float m_new = fmaxf(m_run, tmax);
    float corr = __expf(m_run - m_new);
#pragma unroll
    for (int i = 0; i < 8; ++i) { o0[i] *= corr; o1[i] *= corr; o2[i] *= corr; o3[i] *= corr; }

    const bool hi = lane >= 16;
    float ltot = 0.0f;
    float p0[8], p1[8];
#pragma unroll
    for (int i = 0; i < 8; ++i) {
      p0[i] = __expf(sc[0][i] - m_new);
      p1[i] = __expf(sc[1][i] - m_new);
      ltot += p0[i] + p1[i];
    }
    Frag16 bp0 = build_p_frag(p0, p1, hi);
#pragma unroll
    for (int i = 0; i < 8; ++i) {
      p0[i] = __expf(sc[2][i] - m_new);
      p1[i] = __expf(sc[3][i] - m_new);
      ltot += p0[i] + p1[i];
    }
    Frag16 bp1 = build_p_frag(p0, p1, hi);
    ltot += __shfl_xor(ltot, 16);
    l_run = l_run * corr + ltot;
    m_run = m_new;

    // O^T += V^T(16x32 keys) @ P^T(32x16 queries), 4 d-slabs x 2 key groups.
    o0 = wmma_bf16(load_a_frag(vt + 0 * 16 * 64,      64, lane), bp0, o0);
    o1 = wmma_bf16(load_a_frag(vt + 1 * 16 * 64,      64, lane), bp0, o1);
    o2 = wmma_bf16(load_a_frag(vt + 2 * 16 * 64,      64, lane), bp0, o2);
    o3 = wmma_bf16(load_a_frag(vt + 3 * 16 * 64,      64, lane), bp0, o3);
    o0 = wmma_bf16(load_a_frag(vt + 0 * 16 * 64 + 32, 64, lane), bp1, o0);
    o1 = wmma_bf16(load_a_frag(vt + 1 * 16 * 64 + 32, 64, lane), bp1, o1);
    o2 = wmma_bf16(load_a_frag(vt + 2 * 16 * 64 + 32, 64, lane), bp1, o2);
    o3 = wmma_bf16(load_a_frag(vt + 3 * 16 * 64 + 32, 64, lane), bp1, o3);
    __syncthreads();
  }

  float inv_l = 1.0f / l_run;
  int n = lane & 15, hh = lane >> 4;
  int row = (q0 + n) * CDIM + head * HD;
#pragma unroll
  for (int i = 0; i < 8; ++i) {
    O[row + 0  + hh * 8 + i] = f32_to_bf16(o0[i] * inv_l);
    O[row + 16 + hh * 8 + i] = f32_to_bf16(o1[i] * inv_l);
    O[row + 32 + hh * 8 + i] = f32_to_bf16(o2[i] * inv_l);
    O[row + 48 + hh * 8 + i] = f32_to_bf16(o3[i] * inv_l);
  }
}

// ---------------------------------------------------------------- proj GEMM + residual
__global__ void __launch_bounds__(256) k_gemm_proj(const unsigned short* __restrict__ A,
                                                   const unsigned short* __restrict__ BT,
                                                   const float* __restrict__ X,
                                                   float* __restrict__ Out) {
  __shared__ __align__(16) unsigned short sB[2][64 * 64];
  int tid = threadIdx.x, lane = tid & 31, wv = tid >> 5;
  int sup = blockIdx.x >> 3;                // 0..31
  int nt  = blockIdx.x & 7;                 // 0..7
  int m0 = sup * 128 + wv * 16, n0 = nt * 64;
  const unsigned short* Arow  = A + m0 * CDIM;
  const unsigned short* BTrow = BT + n0 * CDIM;
  v8f acc[4] = {};
  stage_bt_tile64(BTrow, CDIM, sB[0], 0, tid);
  for (int k0 = 0, it = 0; k0 < CDIM; k0 += 64, ++it) {
    int cur = it & 1;
    wait_async();
    __syncthreads();
    if (k0 + 64 < CDIM) stage_bt_tile64(BTrow, CDIM, sB[cur ^ 1], k0 + 64, tid);
    Frag16 af0 = load_a_frag(Arow + k0,      CDIM, lane);
    Frag16 af1 = load_a_frag(Arow + k0 + 32, CDIM, lane);
#pragma unroll
    for (int s = 0; s < 4; ++s) {
      acc[s] = wmma_bf16(af0, load_bt_frag(sB[cur] + s * 16 * 64,      64, lane), acc[s]);
      acc[s] = wmma_bf16(af1, load_bt_frag(sB[cur] + s * 16 * 64 + 32, 64, lane), acc[s]);
    }
    __syncthreads();
  }
  int n = lane & 15, hh = lane >> 4;
#pragma unroll
  for (int s = 0; s < 4; ++s)
#pragma unroll
    for (int i = 0; i < 8; ++i) {
      int idx = (m0 + hh * 8 + i) * CDIM + n0 + s * 16 + n;
      Out[idx] = X[idx] + acc[s][i];
    }
}

extern "C" void kernel_launch(void* const* d_in, const int* in_sizes, int n_in,
                              void* d_out, int out_size, void* d_ws, size_t ws_size,
                              hipStream_t stream) {
  const float* x      = (const float*)d_in[0];
  const float* freqs  = (const float*)d_in[1];
  const float* ln_g   = (const float*)d_in[2];
  const float* ln_b   = (const float*)d_in[3];
  const float* w_qkv  = (const float*)d_in[4];
  const float* w_proj = (const float*)d_in[5];
  float* out = (float*)d_out;

  char* ws = (char*)d_ws;
  unsigned short* h    = (unsigned short*)(ws + (0ull  << 20));  // 4096x512  bf16  (4 MiB)
  unsigned short* wtq  = (unsigned short*)(ws + (4ull  << 20));  // 1536x512  bf16  (1.5 MiB)
  unsigned short* wtp  = (unsigned short*)(ws + (6ull  << 20));  // 512x512   bf16  (0.5 MiB)
  float*          qkv  = (float*)         (ws + (8ull  << 20));  // 4096x1536 f32   (24 MiB)
  unsigned short* qb   = (unsigned short*)(ws + (32ull << 20));  // [8][4096][64]   (4 MiB)
  unsigned short* kb   = (unsigned short*)(ws + (36ull << 20));  // [8][4096][64]   (4 MiB)
  unsigned short* vt   = (unsigned short*)(ws + (40ull << 20));  // [8][64][4096]   (4 MiB)
  unsigned short* ao   = (unsigned short*)(ws + (44ull << 20));  // 4096x512  bf16  (4 MiB)

  k_w_bf16_t  <<<dim3(3072), dim3(256), 0, stream>>>(w_qkv,  wtq, 512, 1536);
  k_w_bf16_t  <<<dim3(1024), dim3(256), 0, stream>>>(w_proj, wtp, 512, 512);
  k_layernorm <<<dim3(4096), dim3(256), 0, stream>>>(x, ln_g, ln_b, h);
  k_gemm_qkv  <<<dim3(768),  dim3(256), 0, stream>>>(h, wtq, qkv);
  k_rope_split<<<dim3(4096), dim3(256), 0, stream>>>(qkv, freqs, qb, kb, vt);
  k_attn      <<<dim3(256),  dim3(256), 0, stream>>>(qb, kb, vt, ao);
  k_gemm_proj <<<dim3(256),  dim3(256), 0, stream>>>(ao, wtp, x, out);
}
